// SelfAttnModel_44212393345623
// MI455X (gfx1250) — matrix operand order repaired and checked by reading, not compiled, not run on hardware
//
#include <hip/hip_runtime.h>

typedef float    v2f  __attribute__((ext_vector_type(2)));
typedef float    v4f  __attribute__((ext_vector_type(4)));
typedef float    v8f  __attribute__((ext_vector_type(8)));
typedef _Float16 v16h __attribute__((ext_vector_type(16)));

#define N_PIX   4096
#define C_DIM   256
#define CQK_D   32
#define ESTRIDE 4100   // 4096 + 4 pad: stride mod 64 banks = 4 -> conflict-free column reads
#define QSTRIDE 34

// ---------------------------------------------------------------------------
// Kernel 1: fused QKV 1x1-conv projections.
//   x (16384 x 256) @ {Wq,Wk: 256x32, Wv: 256x256}, f32 WMMA 16x16x4.
//   q,k stored f32 row-major (B*N, 32); v stored f16 TRANSPOSED (B, C, N) so
//   the A@V GEMM's B-fragments are contiguous 32B loads.
// grid = (1024 row-tiles, 20 col-tiles[2 q | 2 k | 16 v]), block = 1 wave.
// ---------------------------------------------------------------------------
__global__ __launch_bounds__(32) void qkv_proj_kernel(
    const float* __restrict__ x,
    const float* __restrict__ Wq, const float* __restrict__ bq,
    const float* __restrict__ Wk, const float* __restrict__ bk,
    const float* __restrict__ Wv, const float* __restrict__ bv,
    float* __restrict__ qws, float* __restrict__ kws, _Float16* __restrict__ vws)
{
    const int lane = threadIdx.x;
    const int rt   = blockIdx.x;          // 16-row tile of the 16384 pixel rows
    const int yt   = blockIdx.y;          // which 16-col tile of [q|k|v]

    const float* W; const float* bias; int ncols; int col0;
    if (yt < 2)      { W = Wq; bias = bq; ncols = CQK_D; col0 = yt * 16; }
    else if (yt < 4) { W = Wk; bias = bk; ncols = CQK_D; col0 = (yt - 2) * 16; }
    else             { W = Wv; bias = bv; ncols = C_DIM; col0 = (yt - 4) * 16; }

    const int mrow = lane & 15;           // A: row m ; B/D: col n  (both l%16)
    const int half = lane >> 4;           // K split across lane halves
    const int n    = col0 + mrow;
    const size_t xrow = (size_t)(rt * 16 + mrow) * C_DIM;

    v8f acc = {};
    for (int kk = 0; kk < C_DIM; kk += 4) {
        const int kb = kk + half * 2;
        v2f a;  a.x  = x[xrow + kb];                 a.y  = x[xrow + kb + 1];
        v2f bm; bm.x = W[(size_t)kb * ncols + n];    bm.y = W[(size_t)(kb + 1) * ncols + n];
        acc = __builtin_amdgcn_wmma_f32_16x16x4_f32(false, a, false, bm,
                                                    (short)0, acc, false, false);
    }

    const float bb    = bias[n];
    const int   mbase = half * 8;         // D: lanes 0-15 rows 0-7, lanes 16-31 rows 8-15
    if (yt < 4) {
        float* outp = (yt < 2) ? qws : kws;
        #pragma unroll
        for (int r = 0; r < 8; ++r) {
            const int g = rt * 16 + mbase + r;
            outp[(size_t)g * CQK_D + n] = acc[r] + bb;
        }
    } else {
        #pragma unroll
        for (int r = 0; r < 8; ++r) {
            const int g = rt * 16 + mbase + r;            // g = b*4096 + pixel
            vws[((size_t)(g >> 12) * C_DIM + n) * N_PIX + (g & (N_PIX - 1))]
                = (_Float16)(acc[r] + bb);
        }
    }
}

// ---------------------------------------------------------------------------
// Kernel 2: fused attention for one (batch, 16-row tile).
//   Phase 1: energy E[16][4096] = q kT (f32 WMMA) kept in LDS.
//   Phase 2: row softmax in f32, vectorized x4; attention hits HBM once.
//   Phase 3: out = A @ V with f16 WMMA (f32 accum), shared A-fragment across
//            two accumulators per wave, + gamma*(.) + x residual.
// grid = B*256 = 1024 workgroups, block = 256 threads (8 waves).
// Dynamic LDS: 16*ESTRIDE + 16*QSTRIDE floats = 264,576 B (< 320 KB/WGP).
// ---------------------------------------------------------------------------
__global__ __launch_bounds__(256) void attn_fused_kernel(
    const float* __restrict__ qws, const float* __restrict__ kws,
    const _Float16* __restrict__ vws, const float* __restrict__ x,
    const float* __restrict__ gamma_p,
    float* __restrict__ out, float* __restrict__ att)
{
    extern __shared__ float smem[];
    float* Esh = smem;                    // [16][ESTRIDE] energy -> attention
    float* qsh = smem + 16 * ESTRIDE;     // [16][QSTRIDE] q tile

    const int tid  = threadIdx.x;
    const int lane = tid & 31;
    const int w    = tid >> 5;            // wave id 0..7
    const int b    = blockIdx.x >> 8;
    const int i0   = (blockIdx.x & 255) * 16;

    // stage q tile (16x32) into LDS
    for (int idx = tid; idx < 16 * CQK_D; idx += 256) {
        const int r = idx >> 5, c = idx & 31;
        qsh[r * QSTRIDE + c] = qws[((size_t)b * N_PIX + i0 + r) * CQK_D + c];
    }
    __syncthreads();

    const int mrow  = lane & 15;
    const int half  = lane >> 4;
    const int mbase = half * 8;

    // ---- Phase 1: energy tiles, waves stride over 256 j-tiles ----
    for (int jt = w; jt < 256; jt += 8) {
        const int j0 = jt * 16;
        const float* krow = kws + ((size_t)b * N_PIX + j0 + mrow) * CQK_D; // kT col j
        v8f acc = {};
        for (int kk = 0; kk < CQK_D; kk += 4) {
            const int kb = kk + half * 2;
            v2f a;  a.x  = qsh[mrow * QSTRIDE + kb]; a.y  = qsh[mrow * QSTRIDE + kb + 1];
            v2f bm; bm.x = krow[kb];                 bm.y = krow[kb + 1];
            acc = __builtin_amdgcn_wmma_f32_16x16x4_f32(false, a, false, bm,
                                                        (short)0, acc, false, false);
        }
        #pragma unroll
        for (int r = 0; r < 8; ++r)
            Esh[(mbase + r) * ESTRIDE + j0 + mrow] = acc[r];
    }
    __syncthreads();

    // ---- Phase 2: f32 softmax, 2 rows per wave, vectorized x4 ----
    for (int rr = 0; rr < 2; ++rr) {
        const int row = w * 2 + rr;
        float* E = Esh + row * ESTRIDE;
        float mx = -3.402823466e38f;
        for (int j = lane * 4; j < N_PIX; j += 128) {
            const v4f e = *(const v4f*)(E + j);
            mx = fmaxf(fmaxf(fmaxf(mx, e.x), fmaxf(e.y, e.z)), e.w);
        }
        for (int off = 16; off >= 1; off >>= 1) mx = fmaxf(mx, __shfl_xor(mx, off, 32));
        float s = 0.f;
        for (int j = lane * 4; j < N_PIX; j += 128) {
            v4f e = *(const v4f*)(E + j);
            e.x = __expf(e.x - mx); e.y = __expf(e.y - mx);
            e.z = __expf(e.z - mx); e.w = __expf(e.w - mx);
            *(v4f*)(E + j) = e;
            s += (e.x + e.y) + (e.z + e.w);
        }
        for (int off = 16; off >= 1; off >>= 1) s += __shfl_xor(s, off, 32);
        const float inv = 1.0f / s;
        float* arow = att + ((size_t)b * N_PIX + i0 + row) * N_PIX;
        for (int j = lane * 4; j < N_PIX; j += 128) {
            v4f e = *(const v4f*)(E + j);
            e.x *= inv; e.y *= inv; e.z *= inv; e.w *= inv;
            *(v4f*)(E + j) = e;                 // f32 attention for phase 3
            *(v4f*)(arow + j) = e;              // coalesced b128 store to HBM
        }
    }
    __syncthreads();

    // ---- Phase 3: out = A @ V, f16 WMMA 16x16x32, f32 accumulate.
    //      One shared A-fragment feeds two independent accumulators. ----
    const float g = gamma_p[0];
    const int   n0 = w * 32;                                  // cols [n0, n0+32)
    const _Float16* vrow0 = vws + ((size_t)b * C_DIM + n0 + mrow) * N_PIX;      // vT
    const _Float16* vrow1 = vrow0 + (size_t)16 * N_PIX;
    const float*    Erow  = Esh + mrow * ESTRIDE;
    v8f acc0 = {}, acc1 = {};
    for (int kk = 0; kk < N_PIX; kk += 32) {
        // prefetch the vT stream ~2KB ahead (wrapped to stay in-bounds)
        const int pofs = (kk + 1024) & (N_PIX - 1);
        __builtin_prefetch(vrow0 + pofs, 0, 0);
        __builtin_prefetch(vrow1 + pofs, 0, 0);

        // A 16x32 f16 fragment: one contiguous 16-float chunk, cvt to f16
        //   af[0..7]  = Erow[kk + 16*half + 0..7]
        //   af[8..15] = Erow[kk + 16*half + 8 + 0..7]
        const float* pa0 = Erow + kk + half * 16;
        v16h af;
        #pragma unroll
        for (int i = 0; i < 8; ++i) {
            af[i]     = (_Float16)pa0[i];
            af[8 + i] = (_Float16)pa0[8 + i];
        }
        // B 32x16 f16 fragment: 16 contiguous halves (32B) per lane
        const v16h bf0 = *(const v16h*)(vrow0 + kk + half * 16);
        const v16h bf1 = *(const v16h*)(vrow1 + kk + half * 16);

        acc0 = __builtin_amdgcn_wmma_f32_16x16x32_f16(false, af, false, bf0,
                                                      (short)0, acc0, false, false);
        acc1 = __builtin_amdgcn_wmma_f32_16x16x32_f16(false, af, false, bf1,
                                                      (short)0, acc1, false, false);
    }
    #pragma unroll
    for (int r = 0; r < 8; ++r) {
        const size_t row = (size_t)b * N_PIX + i0 + mbase + r;
        const size_t o0  = row * C_DIM + n0 + mrow;
        const size_t o1  = o0 + 16;
        out[o0] = acc0[r] * g + x[o0];
        out[o1] = acc1[r] * g + x[o1];
    }
}

// ---------------------------------------------------------------------------
extern "C" void kernel_launch(void* const* d_in, const int* in_sizes, int n_in,
                              void* d_out, int out_size, void* d_ws, size_t ws_size,
                              hipStream_t stream)
{
    const float* x  = (const float*)d_in[0];
    const float* Wq = (const float*)d_in[1];
    const float* bq = (const float*)d_in[2];
    const float* Wk = (const float*)d_in[3];
    const float* bk = (const float*)d_in[4];
    const float* Wv = (const float*)d_in[5];
    const float* bv = (const float*)d_in[6];
    const float* gm = (const float*)d_in[7];

    float* out = (float*)d_out;
    float* att = out + (size_t)4 * N_PIX * C_DIM;   // outputs: [out | attention]

    // workspace: q (2MB f32) | k (2MB f32) | vT (8MB f16)  -> 12 MB total
    float*    qws = (float*)d_ws;
    float*    kws = qws + (size_t)4 * N_PIX * CQK_D;
    _Float16* vws = (_Float16*)(kws + (size_t)4 * N_PIX * CQK_D);

    dim3 g1(1024, 20, 1);
    qkv_proj_kernel<<<g1, 32, 0, stream>>>(x, Wq, bq, Wk, bk, Wv, bv, qws, kws, vws);

    const size_t shbytes = (size_t)(16 * ESTRIDE + 16 * QSTRIDE) * sizeof(float);
    attn_fused_kernel<<<dim3(1024), dim3(256), shbytes, stream>>>(
        qws, kws, vws, x, gm, out, att);
}
